// HeterogeneousGAT_49426483642740
// MI455X (gfx1250) — matrix operand order repaired and checked by reading, not compiled, not run on hardware
//
#include <hip/hip_runtime.h>
#include <hip/hip_bf16.h>

typedef __attribute__((ext_vector_type(16))) _Float16 v16h;
typedef __attribute__((ext_vector_type(8)))  float    v8f;

#define WMMA16(a,b,c) __builtin_amdgcn_wmma_f32_16x16x32_f16(false,(a),false,(b),(short)0,(c),false,false)

__device__ __forceinline__ float eluf(float x){ return x > 0.f ? x : (expf(x) - 1.f); }
__device__ __forceinline__ float lreluf(float x){ return x > 0.f ? x : 0.2f * x; }

// Fragment-major LDS offset helpers (one 32-K block = 512 halves = 32 lanes x 16).
// A-matrix 16x32 f16 layout: lane(hi*16+row) element e holds K=(e<8?e:e+8)+hi*8.
__device__ __forceinline__ int afrag_off(int kk, int row){
  int hi = (kk >> 3) & 1;
  int e  = (kk & 7) | ((kk >> 4) << 3);
  return (hi*16 + row)*16 + e;
}
// B-matrix 32x16 f16 layout: lane(hi*16+col) element e holds K=(e<8?e:e+8)+hi*8.
__device__ __forceinline__ int bfrag_off(int kk, int col16){
  int hi = (kk >> 3) & 1;
  int e  = (kk & 7) | ((kk >> 4) << 3);
  return (hi*16 + col16)*16 + e;
}

__device__ __forceinline__ void atomicMaxF(float* addr, float val){
  unsigned int* ua = (unsigned int*)addr;
  unsigned int old = *ua, assumed;
  while (__uint_as_float(old) < val){
    assumed = old;
    old = atomicCAS(ua, assumed, __float_as_uint(val));
    if (old == assumed) break;
  }
}

// ---------------------------------------------------------------------------
// Small projection: Y(n x 8) = X(n x k, row-stride ldx) @ W(k x 8)   (k <= 8)
// ---------------------------------------------------------------------------
__global__ void proj8_kernel(const float* __restrict__ X, long n, int k, int ldx,
                             const float* __restrict__ W, float* __restrict__ Y){
  long i = (long)blockIdx.x * blockDim.x + threadIdx.x;
  if (i >= n) return;
  float xv[8];
  #pragma unroll
  for (int t = 0; t < 8; t++) xv[t] = (t < k) ? X[i*ldx + t] : 0.f;
  #pragma unroll
  for (int j = 0; j < 8; j++){
    float s = 0.f;
    for (int t = 0; t < k; t++) s += xv[t] * W[t*8 + j];
    Y[i*8 + j] = s;
  }
}

// ---------------------------------------------------------------------------
// Attention scores
// ---------------------------------------------------------------------------
__global__ void sa_kernel(const float* __restrict__ r, long n,
                          const float* __restrict__ a_self, float* __restrict__ sa){
  long i = (long)blockIdx.x * blockDim.x + threadIdx.x;
  if (i >= n) return;
  float s = 0.f;
  #pragma unroll
  for (int j = 0; j < 8; j++) s += r[i*8 + j] * (a_self[j] + a_self[8 + j]);
  sa[i] = lreluf(s);
}

__global__ void ca_kernel(const float* __restrict__ r, const float* __restrict__ o,
                          const int* __restrict__ req0, const int* __restrict__ req1,
                          long ne, const float* __restrict__ a_cross, float* __restrict__ ca){
  long e = (long)blockIdx.x * blockDim.x + threadIdx.x;
  if (e >= ne) return;
  long ri = req1[e], oi = req0[e];
  float s = 0.f;
  #pragma unroll
  for (int j = 0; j < 8; j++)
    s += r[ri*8 + j] * a_cross[j] + o[oi*8 + j] * a_cross[8 + j];
  ca[e] = lreluf(s);
}

// ---------------------------------------------------------------------------
// Reductions for the joint softmax
// ---------------------------------------------------------------------------
__global__ void set2_kernel(float* p, float a, float b){
  if (threadIdx.x == 0){ p[0] = a; p[1] = b; }
}

__global__ void reduce_max_kernel(const float* __restrict__ x, long n, float* out){
  __shared__ float sh[256];
  float m = -3.0e38f;
  for (long i = (long)blockIdx.x * blockDim.x + threadIdx.x; i < n;
       i += (long)gridDim.x * blockDim.x) m = fmaxf(m, x[i]);
  sh[threadIdx.x] = m; __syncthreads();
  for (int s = 128; s > 0; s >>= 1){
    if (threadIdx.x < s) sh[threadIdx.x] = fmaxf(sh[threadIdx.x], sh[threadIdx.x + s]);
    __syncthreads();
  }
  if (threadIdx.x == 0) atomicMaxF(out, sh[0]);
}

// MZ[0] = max (input), MZ[1] += sum exp(x - max)
__global__ void reduce_sumexp_kernel(const float* __restrict__ x, long n, float* MZ){
  __shared__ float sh[256];
  float M = MZ[0], s = 0.f;
  for (long i = (long)blockIdx.x * blockDim.x + threadIdx.x; i < n;
       i += (long)gridDim.x * blockDim.x) s += expf(x[i] - M);
  sh[threadIdx.x] = s; __syncthreads();
  for (int t = 128; t > 0; t >>= 1){
    if (threadIdx.x < t) sh[threadIdx.x] += sh[threadIdx.x + t];
    __syncthreads();
  }
  if (threadIdx.x == 0) atomicAdd(&MZ[1], sh[0]);
}

// ---------------------------------------------------------------------------
// Edge aggregation (segment sums via global float atomics)
// ---------------------------------------------------------------------------
__global__ void agg_res_kernel(const float* __restrict__ ca, const float* __restrict__ o,
                               const int* __restrict__ req0, const int* __restrict__ req1,
                               long ne, const float* __restrict__ MZ, float* __restrict__ agg){
  long e = (long)blockIdx.x * blockDim.x + threadIdx.x;
  if (e >= ne) return;
  float wgt = expf(ca[e] - MZ[0]) / MZ[1];
  long ri = req1[e], oi = req0[e];
  #pragma unroll
  for (int j = 0; j < 8; j++) atomicAdd(&agg[ri*8 + j], wgt * o[oi*8 + j]);
}

__global__ void res_update_kernel(const float* __restrict__ r, const float* __restrict__ sa,
                                  const float* __restrict__ agg, const float* __restrict__ MZ,
                                  long n, float* __restrict__ out){
  long i = (long)blockIdx.x * blockDim.x + threadIdx.x;
  if (i >= n) return;
  float ns = expf(sa[i] - MZ[0]) / MZ[1];
  #pragma unroll
  for (int j = 0; j < 8; j++) out[i*8 + j] = eluf(ns * r[i*8 + j] + agg[i*8 + j]);
}

__global__ void agg_ops_kernel(const float* __restrict__ res,
                               const int* __restrict__ req0, const int* __restrict__ req1,
                               long ne, float* __restrict__ agg){
  long e = (long)blockIdx.x * blockDim.x + threadIdx.x;
  if (e >= ne) return;
  long oi = req0[e], ri = req1[e];
  #pragma unroll
  for (int j = 0; j < 8; j++) atomicAdd(&agg[oi*8 + j], res[ri*8 + j]);
}

__global__ void prec_kernel(const float* __restrict__ ops, int ld, int dim,
                            const int* __restrict__ src, const int* __restrict__ dst, long ne,
                            float* psum, float* pcnt, float* ssum, float* scnt){
  long e = (long)blockIdx.x * blockDim.x + threadIdx.x;
  if (e >= ne) return;
  long s = src[e], d = dst[e];
  for (int j = 0; j < dim; j++){
    atomicAdd(&psum[d*8 + j], ops[s*ld + j]);
    atomicAdd(&ssum[s*8 + j], ops[d*ld + j]);
  }
  atomicAdd(&pcnt[d], 1.f);
  atomicAdd(&scnt[s], 1.f);
}

__global__ void normcnt_kernel(float* psum, const float* pcnt,
                               float* ssum, const float* scnt, long n){
  long i = (long)blockIdx.x * blockDim.x + threadIdx.x;
  if (i >= n) return;
  float pc = fmaxf(pcnt[i], 1.f), sc = fmaxf(scnt[i], 1.f);
  #pragma unroll
  for (int j = 0; j < 8; j++){ psum[i*8 + j] /= pc; ssum[i*8 + j] /= sc; }
}

__global__ void concat4_kernel(const float* a, const float* b, const float* c,
                               const float* d, long n, float* __restrict__ y){
  long i = (long)blockIdx.x * blockDim.x + threadIdx.x;
  if (i >= n) return;
  #pragma unroll
  for (int j = 0; j < 8; j++){
    y[i*32 + j]      = a[i*8 + j];
    y[i*32 + 8 + j]  = b[i*8 + j];
    y[i*32 + 16 + j] = c[i*8 + j];
    y[i*32 + 24 + j] = d[i*8 + j];
  }
}

__global__ void mask_ends_kernel(float* y, long n){
  int j = threadIdx.x;
  if (j < 8){ y[j] = 0.f; y[(n - 1)*8 + j] = 0.f; }
}

__global__ void mean8_kernel(const float* __restrict__ X, long n, float inv, float* out){
  __shared__ float sh[8];
  if (threadIdx.x < 8) sh[threadIdx.x] = 0.f;
  __syncthreads();
  long i = (long)blockIdx.x * blockDim.x + threadIdx.x;
  if (i < n){
    #pragma unroll
    for (int j = 0; j < 8; j++) atomicAdd(&sh[j], X[i*8 + j]);   // ds_add_f32
  }
  __syncthreads();
  if (threadIdx.x < 8) atomicAdd(&out[threadIdx.x], sh[threadIdx.x] * inv);
}

// ---------------------------------------------------------------------------
// Fused 3-layer MLP  [in_dim -> 128 -> 128 -> 8], elu on hidden, via WMMA f16.
// Block = 128 threads (4 waves); each wave owns a 16-row tile (64 rows/block).
// All LDS buffers are stored in fragment-major order: each WMMA operand is a
// single aligned 32-byte v16h load per lane.
// ---------------------------------------------------------------------------
__global__ __launch_bounds__(128)
void fused_mlp3_kernel(const float* __restrict__ X, long n, int in_dim, int ldx,
                       const float* __restrict__ W1, const float* __restrict__ B1,
                       const float* __restrict__ W2, const float* __restrict__ B2,
                       const float* __restrict__ W3, const float* __restrict__ B3,
                       float* __restrict__ Y){
  __shared__ __align__(32) _Float16 w1h[8*512];      // 8 N-tiles, K padded to 32
  __shared__ __align__(32) _Float16 w2h[4*8*512];    // 4 K-blocks x 8 N-tiles
  __shared__ __align__(32) _Float16 w3h[4*512];      // 4 K-blocks, N padded to 16
  __shared__ float b1s[128], b2s[128], b3s[8];
  __shared__ __align__(32) _Float16 hst[4][4*512];   // per-wave 16x128, A-frag order

  const int t = threadIdx.x;
  for (int i = t; i < 32*128; i += 128){
    int k = i >> 7, c = i & 127;
    _Float16 v = (k < in_dim) ? (_Float16)W1[k*128 + c] : (_Float16)0.f;
    w1h[(c >> 4)*512 + bfrag_off(k, c & 15)] = v;
  }
  for (int i = t; i < 128*128; i += 128){
    int k = i >> 7, c = i & 127;
    w2h[((k >> 5)*8 + (c >> 4))*512 + bfrag_off(k & 31, c & 15)] = (_Float16)W2[i];
  }
  for (int i = t; i < 128*16; i += 128){
    int k = i >> 4, c = i & 15;
    _Float16 v = (c < 8) ? (_Float16)W3[k*8 + c] : (_Float16)0.f;
    w3h[(k >> 5)*512 + bfrag_off(k & 31, c)] = v;
  }
  if (t < 128){ b1s[t] = B1[t]; b2s[t] = B2[t]; }
  if (t < 8) b3s[t] = B3[t];
  __syncthreads();

  const int wave = t >> 5, lane = t & 31, hi = lane >> 4, l15 = lane & 15;
  const long row0 = (long)blockIdx.x * 64 + wave * 16;
  long rA = row0 + l15; if (rA >= n) rA = n - 1;   // clamp; padded rows never stored

  // ---- layer 1: (16 x in) @ (in x 128) ----
  v16h a1;
  #pragma unroll
  for (int e = 0; e < 16; e++){
    int K = (e < 8 ? e : e + 8) + hi*8;
    a1[e] = (K < in_dim) ? (_Float16)X[rA*ldx + K] : (_Float16)0.f;
  }
  v8f acc[8];
  #pragma unroll
  for (int nt = 0; nt < 8; nt++){
    v16h b = *(const v16h*)&w1h[nt*512 + lane*16];
    v8f z = {};
    acc[nt] = WMMA16(a1, b, z);
  }
  #pragma unroll
  for (int nt = 0; nt < 8; nt++)
    #pragma unroll
    for (int v = 0; v < 8; v++){
      int col = nt*16 + l15, mrow = v + hi*8;
      hst[wave][(col >> 5)*512 + afrag_off(col & 31, mrow)] =
          (_Float16)eluf(acc[nt][v] + b1s[col]);
    }

  // ---- layer 2: (16 x 128) @ (128 x 128) ----
  v8f acc2[8];
  #pragma unroll
  for (int nt = 0; nt < 8; nt++){ v8f z = {}; acc2[nt] = z; }
  #pragma unroll
  for (int kb = 0; kb < 4; kb++){
    v16h a2 = *(const v16h*)&hst[wave][kb*512 + lane*16];
    #pragma unroll
    for (int nt = 0; nt < 8; nt++){
      v16h b = *(const v16h*)&w2h[(kb*8 + nt)*512 + lane*16];
      acc2[nt] = WMMA16(a2, b, acc2[nt]);
    }
  }
  #pragma unroll
  for (int nt = 0; nt < 8; nt++)
    #pragma unroll
    for (int v = 0; v < 8; v++){
      int col = nt*16 + l15, mrow = v + hi*8;
      hst[wave][(col >> 5)*512 + afrag_off(col & 31, mrow)] =
          (_Float16)eluf(acc2[nt][v] + b2s[col]);
    }

  // ---- layer 3: (16 x 128) @ (128 x 8-padded-16), no activation ----
  v8f acc3 = {};
  #pragma unroll
  for (int kb = 0; kb < 4; kb++){
    v16h a3 = *(const v16h*)&hst[wave][kb*512 + lane*16];
    v16h b  = *(const v16h*)&w3h[kb*512 + lane*16];
    acc3 = WMMA16(a3, b, acc3);
  }
  #pragma unroll
  for (int v = 0; v < 8; v++){
    long row = row0 + v + hi*8;
    if (row < n && l15 < 8) Y[row*8 + l15] = acc3[v] + b3s[l15];
  }
}

// ---------------------------------------------------------------------------
// Generic WMMA GEMM:  Y(n x m) = act(X(n x K) @ W(K x m) + bias)
// grid (ceil(n/16), ceil(m/128)); block 256 (8 waves, one 16x16 tile each).
// A/B staged in LDS in fragment-major order (vector ds loads per operand).
// ---------------------------------------------------------------------------
__global__ __launch_bounds__(256)
void gemm_wmma_kernel(const float* __restrict__ X, long n, long K,
                      const float* __restrict__ W, long m,
                      const float* __restrict__ bias, float* __restrict__ Y, int act){
  __shared__ __align__(32) _Float16 As[512];        // 16x32 A-frag order
  __shared__ __align__(32) _Float16 Bs[8*512];      // 8 col-tiles x 32x16 B-frag order
  const int t = threadIdx.x, wave = t >> 5, lane = t & 31, hi = lane >> 4, l15 = lane & 15;
  const long rowBase = (long)blockIdx.x * 16;
  const long colBase = (long)blockIdx.y * 128;
  v8f acc = {};
  for (long k0 = 0; k0 < K; k0 += 32){
    __syncthreads();
    for (int i = t; i < 512; i += 256){
      int rr = i >> 5, kk = i & 31;
      long gr = rowBase + rr, gk = k0 + kk;
      _Float16 v = (gr < n && gk < K) ? (_Float16)X[gr*K + gk] : (_Float16)0.f;
      As[afrag_off(kk, rr)] = v;
    }
    for (int i = t; i < 4096; i += 256){
      int kk = i >> 7, cc = i & 127;
      long gk = k0 + kk, gc = colBase + cc;
      _Float16 v = (gk < K && gc < m) ? (_Float16)W[gk*m + gc] : (_Float16)0.f;
      Bs[(cc >> 4)*512 + bfrag_off(kk, cc & 15)] = v;
    }
    if (k0 + 32 < K && t == 0) __builtin_prefetch(&W[(k0 + 32)*m + colBase], 0, 1);
    __syncthreads();
    v16h a = *(const v16h*)&As[lane*16];
    v16h b = *(const v16h*)&Bs[wave*512 + lane*16];
    acc = WMMA16(a, b, acc);
  }
  long col = colBase + wave*16 + l15;
  #pragma unroll
  for (int v = 0; v < 8; v++){
    long row = rowBase + v + hi*8;
    if (row < n && col < m){
      float x = acc[v] + bias[col];
      if (act == 2) x = tanhf(x);
      Y[row*m + col] = x;
    }
  }
}

// ---------------------------------------------------------------------------
// Actor/critic helpers
// ---------------------------------------------------------------------------
__global__ void actin_kernel(const float* __restrict__ ops, const float* __restrict__ res,
                             const float* __restrict__ gs, const int* __restrict__ act,
                             long na, float* __restrict__ y){
  long i = (long)blockIdx.x * blockDim.x + threadIdx.x;
  if (i >= na) return;
  long a0 = act[i*2], a1 = act[i*2 + 1];
  #pragma unroll
  for (int j = 0; j < 8; j++){
    y[i*32 + j]     = ops[a0*8 + j];
    y[i*32 + 8 + j] = res[a1*8 + j];
  }
  #pragma unroll
  for (int j = 0; j < 16; j++) y[i*32 + 16 + j] = gs[j];
}

__global__ void vecmlp_kernel(const float* __restrict__ x, int nin,
                              const float* __restrict__ W, const float* __restrict__ b,
                              int nout, float* __restrict__ y, int act){
  int j = blockIdx.x * blockDim.x + threadIdx.x;
  if (j >= nout) return;
  float s = b[j];
  for (int i = 0; i < nin; i++) s += x[i] * W[(long)i*nout + j];
  y[j] = (act == 2) ? tanhf(s) : s;
}

__global__ void rowdot_kernel(const float* __restrict__ X, int k,
                              const float* __restrict__ w, const float* __restrict__ b,
                              float* __restrict__ y){
  __shared__ float sh[256];
  long row = blockIdx.x;
  float s = 0.f;
  for (int i = threadIdx.x; i < k; i += blockDim.x) s += X[row*k + i] * w[i];
  sh[threadIdx.x] = s; __syncthreads();
  for (int t = 128; t > 0; t >>= 1){
    if (threadIdx.x < t) sh[threadIdx.x] += sh[threadIdx.x + t];
    __syncthreads();
  }
  if (threadIdx.x == 0) y[row] = sh[0] + b[0];
}

__global__ void softmax_write_kernel(const float* __restrict__ logits, long n,
                                     const float* __restrict__ MZ, float* __restrict__ out){
  long i = (long)blockIdx.x * blockDim.x + threadIdx.x;
  if (i >= n) return;
  out[i] = expf(logits[i] - MZ[0]) / MZ[1];
}

// ---------------------------------------------------------------------------
// Host launcher
// ---------------------------------------------------------------------------
extern "C" void kernel_launch(void* const* d_in, const int* in_sizes, int n_in,
                              void* d_out, int out_size, void* d_ws, size_t ws_size,
                              hipStream_t stream){
  const long N_OPS  = in_sizes[0] / 6;
  const long N_RES  = in_sizes[1] / 3;
  const long N_PREC = in_sizes[2] / 2;
  const long N_EDG  = in_sizes[3] / 2;
  const long N_ACT  = in_sizes[4] / 2;
  const int  AC     = (n_in > 74) ? in_sizes[74] : 4096;   // actor hidden width

  const float* OPS0 = (const float*)d_in[0];
  const float* RES0 = (const float*)d_in[1];
  const int* PSRC = (const int*)d_in[2];
  const int* PDST = PSRC + N_PREC;
  const int* REQ0 = (const int*)d_in[3];
  const int* REQ1 = REQ0 + N_EDG;
  const int* ACTN = (const int*)d_in[4];
  auto P = [&](int i)->const float* { return (const float*)d_in[i]; };

  // workspace carve-out
  float* w = (float*)d_ws;
  auto take = [&](long nfl)->float* { float* p = w; w += nfl; return p; };
  float* ops_a   = take(N_OPS*8);
  float* ops_b   = take(N_OPS*8);
  float* res_a   = take(N_RES*8);
  float* res_b   = take(N_RES*8);
  float* r_      = take(N_RES*8);
  float* o_      = take(N_OPS*8);
  float* sa_     = take(N_RES);
  float* ca_     = take(N_EDG);
  float* agg_res = take(N_RES*8);
  float* agg_ops = take(N_OPS*8);
  float* psum    = take(N_OPS*8);
  float* ssum    = take(N_OPS*8);
  float* pcnt    = take(N_OPS);
  float* scnt    = take(N_OPS);
  float* pe_     = take(N_OPS*8);
  float* se_     = take(N_OPS*8);
  float* sm_     = take(N_OPS*8);
  float* re_     = take(N_OPS*8);
  float* comb_in = take(N_OPS*32);
  float* gs      = take(16);
  float* scalars = take(16);          // [0..1]=attn M,Z  [2..3]=softmax M,Z
  float* a_in    = take(N_ACT*32);
  float* H1      = take(N_ACT*(long)AC);
  float* H2      = take(N_ACT*(long)AC);
  float* ch1     = take(AC);
  float* ch2     = take(AC);
  float* logits  = take(N_ACT);

  const int T = 256;
  auto G  = [&](long n){ return dim3((unsigned)((n + T - 1) / T)); };
  auto GR = [&](long n){ long b = (n + T - 1) / T; if (b > 2048) b = 2048; return dim3((unsigned)b); };

  const float* ops_in = OPS0; int op_ld = 6, op_dim = 6;
  const float* res_in = RES0; int res_ld = 3, res_dim = 3;
  float* ops_bufs[2] = { ops_a, ops_b };
  float* res_bufs[2] = { res_a, res_b };

  for (int l = 0; l < 2; l++){
    const float* Wr  = P(5 + 4*l + 0);
    const float* Wo  = P(5 + 4*l + 1);
    const float* Asf = P(5 + 4*l + 2);
    const float* Acr = P(5 + 4*l + 3);
    const int ob = 13 + 30*l;   // op-layer params base

    // ---- res layer ----
    proj8_kernel<<<G(N_RES), T, 0, stream>>>(res_in, N_RES, res_dim, res_ld, Wr, r_);
    proj8_kernel<<<G(N_OPS), T, 0, stream>>>(ops_in, N_OPS, op_dim, op_ld, Wo, o_);
    set2_kernel<<<1, 32, 0, stream>>>(scalars, -3.0e38f, 0.f);
    sa_kernel<<<G(N_RES), T, 0, stream>>>(r_, N_RES, Asf, sa_);
    ca_kernel<<<G(N_EDG), T, 0, stream>>>(r_, o_, REQ0, REQ1, N_EDG, Acr, ca_);
    reduce_max_kernel<<<GR(N_RES), T, 0, stream>>>(sa_, N_RES, scalars);
    reduce_max_kernel<<<GR(N_EDG), T, 0, stream>>>(ca_, N_EDG, scalars);
    reduce_sumexp_kernel<<<GR(N_RES), T, 0, stream>>>(sa_, N_RES, scalars);
    reduce_sumexp_kernel<<<GR(N_EDG), T, 0, stream>>>(ca_, N_EDG, scalars);
    hipMemsetAsync(agg_res, 0, N_RES*8*sizeof(float), stream);
    agg_res_kernel<<<G(N_EDG), T, 0, stream>>>(ca_, o_, REQ0, REQ1, N_EDG, scalars, agg_res);
    float* res_new = res_bufs[l];
    res_update_kernel<<<G(N_RES), T, 0, stream>>>(r_, sa_, agg_res, scalars, N_RES, res_new);

    // ---- op layer ----
    hipMemsetAsync(agg_ops, 0, N_OPS*8*sizeof(float), stream);
    hipMemsetAsync(psum, 0, N_OPS*8*sizeof(float), stream);
    hipMemsetAsync(ssum, 0, N_OPS*8*sizeof(float), stream);
    hipMemsetAsync(pcnt, 0, N_OPS*sizeof(float), stream);
    hipMemsetAsync(scnt, 0, N_OPS*sizeof(float), stream);
    agg_ops_kernel<<<G(N_EDG), T, 0, stream>>>(res_new, REQ0, REQ1, N_EDG, agg_ops);
    prec_kernel<<<G(N_PREC), T, 0, stream>>>(ops_in, op_ld, op_dim, PSRC, PDST, N_PREC,
                                             psum, pcnt, ssum, scnt);
    normcnt_kernel<<<G(N_OPS), T, 0, stream>>>(psum, pcnt, ssum, scnt, N_OPS);

    dim3 gm((unsigned)((N_OPS + 63) / 64));
    fused_mlp3_kernel<<<gm, 128, 0, stream>>>(psum, N_OPS, op_dim, 8,
        P(ob+0), P(ob+1), P(ob+2), P(ob+3), P(ob+4), P(ob+5), pe_);
    fused_mlp3_kernel<<<gm, 128, 0, stream>>>(ssum, N_OPS, op_dim, 8,
        P(ob+6), P(ob+7), P(ob+8), P(ob+9), P(ob+10), P(ob+11), se_);
    fused_mlp3_kernel<<<gm, 128, 0, stream>>>(ops_in, N_OPS, op_dim, op_ld,
        P(ob+12), P(ob+13), P(ob+14), P(ob+15), P(ob+16), P(ob+17), sm_);
    fused_mlp3_kernel<<<gm, 128, 0, stream>>>(agg_ops, N_OPS, 8, 8,
        P(ob+18), P(ob+19), P(ob+20), P(ob+21), P(ob+22), P(ob+23), re_);
    concat4_kernel<<<G(N_OPS), T, 0, stream>>>(pe_, se_, re_, sm_, N_OPS, comb_in);
    float* ops_new = ops_bufs[l];
    fused_mlp3_kernel<<<gm, 128, 0, stream>>>(comb_in, N_OPS, 32, 32,
        P(ob+24), P(ob+25), P(ob+26), P(ob+27), P(ob+28), P(ob+29), ops_new);
    mask_ends_kernel<<<1, 32, 0, stream>>>(ops_new, N_OPS);

    ops_in = ops_new; op_ld = 8; op_dim = 8;
    res_in = res_new; res_ld = 8; res_dim = 8;
  }

  // ---- graph state ----
  hipMemsetAsync(gs, 0, 16*sizeof(float), stream);
  mean8_kernel<<<G(N_OPS), T, 0, stream>>>(ops_in, N_OPS, 1.f / (float)N_OPS, gs);
  mean8_kernel<<<G(N_RES), T, 0, stream>>>(res_in, N_RES, 1.f / (float)N_RES, gs + 8);

  // ---- critic (params base 79) ----
  vecmlp_kernel<<<G(AC), T, 0, stream>>>(gs, 16, P(79), P(80), AC, ch1, 2);
  vecmlp_kernel<<<G(AC), T, 0, stream>>>(ch1, AC, P(81), P(82), AC, ch2, 2);
  rowdot_kernel<<<1, T, 0, stream>>>(ch2, AC, P(83), P(84), ((float*)d_out) + N_ACT);

  // ---- actor (params base 73) ----
  actin_kernel<<<G(N_ACT), T, 0, stream>>>(ops_in, res_in, gs, ACTN, N_ACT, a_in);
  dim3 gg1((unsigned)((N_ACT + 15) / 16), (unsigned)((AC + 127) / 128));
  gemm_wmma_kernel<<<gg1, 256, 0, stream>>>(a_in, N_ACT, 32, P(73), AC, P(74), H1, 2);
  gemm_wmma_kernel<<<gg1, 256, 0, stream>>>(H1, N_ACT, AC, P(75), AC, P(76), H2, 2);
  rowdot_kernel<<<(unsigned)N_ACT, T, 0, stream>>>(H2, AC, P(77), P(78), logits);
  set2_kernel<<<1, 32, 0, stream>>>(scalars + 2, -3.0e38f, 0.f);
  reduce_max_kernel<<<GR(N_ACT), T, 0, stream>>>(logits, N_ACT, scalars + 2);
  reduce_sumexp_kernel<<<GR(N_ACT), T, 0, stream>>>(logits, N_ACT, scalars + 2);
  softmax_write_kernel<<<G(N_ACT), T, 0, stream>>>(logits, N_ACT, scalars + 2, (float*)d_out);
}
